// GAT_GRAN_26182120636868
// MI455X (gfx1250) — matrix operand order repaired and checked by reading, not compiled, not run on hardware
//
#include <hip/hip_runtime.h>
#include <hip/hip_bf16.h>

// ---------------------------------------------------------------------------
// GAT-GRU message passing, fused for gfx1250 (wave32, WMMA f32_16x16x32_f16).
// Round 3: ALL weight packs staged in LDS (216 KB edge / 256 KB GRU blocks,
// within the 320 KB WGP budget); software-pipelined A/B fragment prefetch.
// ---------------------------------------------------------------------------

typedef _Float16 h4   __attribute__((ext_vector_type(4)));
typedef _Float16 h8   __attribute__((ext_vector_type(8)));
typedef _Float16 v16h __attribute__((ext_vector_type(16)));
typedef float    v8f  __attribute__((ext_vector_type(8)));

#define D_DIM 128
#define E_DIM 32
#define IN_DIM 160              // D + E
#define W1_HALFS (5 * 8 * 512)  // 20480 halfs per 160x128 pack
#define W2_HALFS (4 * 8 * 512)  // 16384 halfs per 128x128 pack
#define WG_HALFS (4 * 24 * 512) // 49152 halfs per 128x384 pack

// ---- A-fragment loader (16-bit A 16x32 layout, per CDNA5 ISA 7.12.2) ------
// lane<16 : M=lane,    elems 0..7 -> K=kb+0..7,  elems 8..15 -> K=kb+16..23
// lane>=16: M=lane-16, elems 0..7 -> K=kb+8..15, elems 8..15 -> K=kb+24..31
__device__ __forceinline__ v16h load_a_frag(const _Float16* base, int rowlen,
                                            int kbase, int lane) {
  const int m    = lane & 15;
  const int koff = kbase + ((lane & 16) ? 8 : 0);
  const _Float16* p = base + m * rowlen + koff;
  h8 lo = *(const h8*)(p);        // K = koff..koff+7   (ds_load_b128)
  h8 hi = *(const h8*)(p + 16);   // K = koff+16..+23
  v16h a;
#pragma unroll
  for (int i = 0; i < 8; ++i) { a[i] = lo[i]; a[8 + i] = hi[i]; }
  return a;
}

// ---- 16xK @ Kx128 accumulation, 1-deep A/B prefetch pipeline --------------
__device__ __forceinline__ void mm_accum(const _Float16* Astage, int rowlen,
                                         int kchunks, const _Float16* Wp,
                                         int lane, v8f acc[8]) {
  v16h a = load_a_frag(Astage, rowlen, 0, lane);
  for (int c = 0; c < kchunks; ++c) {
    v16h an;
    if (c + 1 < kchunks) an = load_a_frag(Astage, rowlen, (c + 1) * 32, lane);
    v16h b = *(const v16h*)(Wp + ((c * 8) * 32 + lane) * 16);
#pragma unroll
    for (int j = 0; j < 8; ++j) {
      v16h bn;
      if (j < 7) bn = *(const v16h*)(Wp + ((c * 8 + j + 1) * 32 + lane) * 16);
      acc[j] = __builtin_amdgcn_wmma_f32_16x16x32_f16(
          false, a, false, b, (short)0, acc[j], false, false);
      if (j < 7) b = bn;
    }
    if (c + 1 < kchunks) a = an;
  }
}

// ---- dual accumulation sharing the A fragment (msg-L1 + att-L1) -----------
__device__ __forceinline__ void mm_accum2(const _Float16* Astage, int rowlen,
                                          int kchunks,
                                          const _Float16* WpM,
                                          const _Float16* WpA,
                                          int lane, v8f accM[8], v8f accA[8]) {
  v16h a = load_a_frag(Astage, rowlen, 0, lane);
  for (int c = 0; c < kchunks; ++c) {
    v16h an;
    if (c + 1 < kchunks) an = load_a_frag(Astage, rowlen, (c + 1) * 32, lane);
    v16h bm = *(const v16h*)(WpM + ((c * 8) * 32 + lane) * 16);
    v16h ba = *(const v16h*)(WpA + ((c * 8) * 32 + lane) * 16);
#pragma unroll
    for (int j = 0; j < 8; ++j) {
      v16h bmn, ban;
      if (j < 7) {
        bmn = *(const v16h*)(WpM + ((c * 8 + j + 1) * 32 + lane) * 16);
        ban = *(const v16h*)(WpA + ((c * 8 + j + 1) * 32 + lane) * 16);
      }
      accM[j] = __builtin_amdgcn_wmma_f32_16x16x32_f16(
          false, a, false, bm, (short)0, accM[j], false, false);
      accA[j] = __builtin_amdgcn_wmma_f32_16x16x32_f16(
          false, a, false, ba, (short)0, accA[j], false, false);
      if (j < 7) { bm = bmn; ba = ban; }
    }
    if (c + 1 < kchunks) a = an;
  }
}

// ---- pack f32 weight (K x Ndim, row-major) into B-fragment layout ---------
// B layout: lane<16 holds K=c*32+{0..7,16..23}, lane>=16 holds K=c*32+{8..15,24..31}, N=j*16+(lane&15)
__global__ void gatgru_pack_w(const float* __restrict__ W,
                              _Float16* __restrict__ out,
                              int kchunks, int ntiles, int Ndim) {
  const int t = blockIdx.x * blockDim.x + threadIdx.x;
  const int total = kchunks * ntiles * 512;
  if (t >= total) return;
  const int e    = t & 15;
  const int lane = (t >> 4) & 31;
  const int f    = t >> 9;
  const int j    = f % ntiles;
  const int c    = f / ntiles;
  const int K    = c * 32 + ((lane & 16) ? 8 : 0) + (e & 7) + ((e & 8) ? 16 : 0);
  const int Nc   = j * 16 + (lane & 15);
  out[t] = (_Float16)W[(size_t)K * Ndim + Nc];
}

__global__ void gatgru_zero(float* __restrict__ p, size_t n4) {
  const size_t i = (size_t)blockIdx.x * blockDim.x + threadIdx.x;
  if (i < n4) {
    float4 z; z.x = 0.f; z.y = 0.f; z.z = 0.f; z.w = 0.f;
    *((float4*)p + i) = z;
  }
}

// ---------------------------------------------------------------------------
// Edge kernel: one wave = 16 edges. gather -> (msg,att) L1 fused -> L2s ->
// gate -> scatter-add. All four weight packs (144 KB) live in LDS.
// ---------------------------------------------------------------------------
__global__ __launch_bounds__(256, 1)
void gatgru_edge_kernel(const float* __restrict__ node_feat,
                        const int*   __restrict__ edge_index,
                        const float* __restrict__ edge_feat,
                        const float* __restrict__ msg_b1,
                        const float* __restrict__ msg_b2,
                        const float* __restrict__ att_b1,
                        const float* __restrict__ att_b2,
                        const _Float16* __restrict__ Wpack, // [mW1|aW1|mW2|aW2]
                        float* __restrict__ accum, int Medges) {
  __shared__ __align__(32) _Float16 WL[2 * W1_HALFS + 2 * W2_HALFS]; // 144 KB
  __shared__ __align__(32) _Float16 S1s[8 * 16 * IN_DIM];            // 40 KB
  __shared__ __align__(32) _Float16 S2s[8 * 16 * D_DIM];             // 32 KB

  // ---- block-cooperative copy of all weight packs into LDS ----
  {
    const float4* src = (const float4*)Wpack;
    float4*       dst = (float4*)WL;
    // 73728 halfs = 147456 B = 9216 float4; 256 threads -> 36 iters
    for (int i = threadIdx.x; i < 9216; i += 256) dst[i] = src[i];
  }
  __syncthreads();
  const _Float16* mW1L = WL;
  const _Float16* aW1L = WL + W1_HALFS;
  const _Float16* mW2L = WL + 2 * W1_HALFS;
  const _Float16* aW2L = WL + 2 * W1_HALFS + W2_HALFS;

  const int lane = threadIdx.x & 31;
  const int w    = threadIdx.x >> 5;
  const int e0   = (blockIdx.x * 8 + w) * 16;
  if (e0 >= Medges) return;                    // wave-uniform (after barrier)
  _Float16* S1 = S1s + w * 16 * IN_DIM;
  _Float16* S2 = S2s + w * 16 * D_DIM;

  // ---- stage edge_input tile [16 x 160] as f16 in LDS ----
  for (int m = 0; m < 16; ++m) {
    const int e = e0 + m;
    const int s = edge_index[e];            // src row
    const int d = edge_index[Medges + e];   // dst row
    const float4 sa = *(const float4*)(node_feat + (size_t)s * D_DIM + lane * 4);
    const float4 da = *(const float4*)(node_feat + (size_t)d * D_DIM + lane * 4);
    h4 hv;
    hv[0] = (_Float16)(sa.x - da.x);
    hv[1] = (_Float16)(sa.y - da.y);
    hv[2] = (_Float16)(sa.z - da.z);
    hv[3] = (_Float16)(sa.w - da.w);
    *(h4*)(S1 + m * IN_DIM + lane * 4) = hv;
    S1[m * IN_DIM + D_DIM + lane] =
        (_Float16)edge_feat[(size_t)e * E_DIM + lane];
  }

  const int nbase = lane & 15;              // C-matrix column within tile
  const int mbase = (lane & 16) ? 8 : 0;    // C-matrix row offset

  // ---- fused layer 1: msg and att share A fragments ----
  v8f accM[8], accA[8];
#pragma unroll
  for (int j = 0; j < 8; ++j) { accM[j] = (v8f)(0.0f); accA[j] = (v8f)(0.0f); }
  mm_accum2(S1, IN_DIM, 5, mW1L, aW1L, lane, accM, accA);

  // bias + relu; msg hidden -> S2, att hidden -> S1 (edge_input now dead)
#pragma unroll
  for (int j = 0; j < 8; ++j) {
    const float bm = msg_b1[j * 16 + nbase];
    const float ba = att_b1[j * 16 + nbase];
#pragma unroll
    for (int r = 0; r < 8; ++r) {
      const int o = (mbase + r) * D_DIM + j * 16 + nbase;
      S2[o] = (_Float16)fmaxf(accM[j][r] + bm, 0.0f);
      S1[o] = (_Float16)fmaxf(accA[j][r] + ba, 0.0f);
    }
  }

  // ---- layer 2 (weights from LDS) ----
  v8f mout[8];
#pragma unroll
  for (int j = 0; j < 8; ++j) mout[j] = (v8f)(0.0f);
  mm_accum(S2, D_DIM, 4, mW2L, lane, mout);

  v8f aout[8];
#pragma unroll
  for (int j = 0; j < 8; ++j) aout[j] = (v8f)(0.0f);
  mm_accum(S1, D_DIM, 4, aW2L, lane, aout);

  // ---- gate + scatter-add into destination nodes ----
  int dsts[8];
#pragma unroll
  for (int r = 0; r < 8; ++r) dsts[r] = edge_index[Medges + e0 + mbase + r];
#pragma unroll
  for (int j = 0; j < 8; ++j) {
    const float mb = msg_b2[j * 16 + nbase];
    const float ab = att_b2[j * 16 + nbase];
#pragma unroll
    for (int r = 0; r < 8; ++r) {
      const float g = 1.0f / (1.0f + __expf(-(aout[j][r] + ab)));
      const float v = (mout[j][r] + mb) * g;
      atomicAdd(accum + (size_t)dsts[r] * D_DIM + j * 16 + nbase, v);
    }
  }
}

// ---------------------------------------------------------------------------
// GRU kernel: one wave = 16 nodes. gi = msg@Wi, gh = h@Wh via WMMA (192/wave),
// weights (192 KB) staged in LDS; elementwise gate combine; store new state.
// ---------------------------------------------------------------------------
__global__ __launch_bounds__(256, 1)
void gatgru_gru_kernel(const float* __restrict__ accum,
                       const float* __restrict__ node_feat,
                       const float* __restrict__ gru_bi,
                       const float* __restrict__ gru_bh,
                       const _Float16* __restrict__ Wpack,  // [Wi|Wh]
                       float* __restrict__ out, int Nn) {
  __shared__ __align__(32) _Float16 WG[2 * WG_HALFS];      // 192 KB
  __shared__ __align__(32) _Float16 SAs[8 * 16 * D_DIM];   // state_msg tile
  __shared__ __align__(32) _Float16 SBs[8 * 16 * D_DIM];   // node_feat tile

  {
    const float4* src = (const float4*)Wpack;
    float4*       dst = (float4*)WG;
    // 98304 halfs = 196608 B = 12288 float4; 256 threads -> 48 iters
    for (int i = threadIdx.x; i < 12288; i += 256) dst[i] = src[i];
  }
  __syncthreads();
  const _Float16* WiL = WG;
  const _Float16* WhL = WG + WG_HALFS;

  const int lane = threadIdx.x & 31;
  const int w    = threadIdx.x >> 5;
  const int nb   = (blockIdx.x * 8 + w) * 16;
  if (nb >= Nn) return;                       // wave-uniform (after barrier)
  _Float16* SA = SAs + w * 16 * D_DIM;
  _Float16* SB = SBs + w * 16 * D_DIM;

  for (int m = 0; m < 16; ++m) {
    const float4 a4 = *(const float4*)(accum + (size_t)(nb + m) * D_DIM + lane * 4);
    const float4 b4 = *(const float4*)(node_feat + (size_t)(nb + m) * D_DIM + lane * 4);
    h4 ha, hb;
    ha[0] = (_Float16)a4.x; ha[1] = (_Float16)a4.y;
    ha[2] = (_Float16)a4.z; ha[3] = (_Float16)a4.w;
    hb[0] = (_Float16)b4.x; hb[1] = (_Float16)b4.y;
    hb[2] = (_Float16)b4.z; hb[3] = (_Float16)b4.w;
    *(h4*)(SA + m * D_DIM + lane * 4) = ha;
    *(h4*)(SB + m * D_DIM + lane * 4) = hb;
  }

  const int nbase = lane & 15;
  const int mbase = (lane & 16) ? 8 : 0;

  for (int t = 0; t < 8; ++t) {             // 8 feature tiles of 16 cols
    v8f ir = (v8f)(0.f), iz = (v8f)(0.f), inn = (v8f)(0.f);
    v8f hr = (v8f)(0.f), hz = (v8f)(0.f), hnn = (v8f)(0.f);
    for (int c = 0; c < 4; ++c) {           // K tiles of 32
      v16h aA = load_a_frag(SA, D_DIM, c * 32, lane);
      v16h aH = load_a_frag(SB, D_DIM, c * 32, lane);
      v16h bir = *(const v16h*)(WiL + ((c * 24 +      t) * 32 + lane) * 16);
      v16h biz = *(const v16h*)(WiL + ((c * 24 +  8 + t) * 32 + lane) * 16);
      v16h bin = *(const v16h*)(WiL + ((c * 24 + 16 + t) * 32 + lane) * 16);
      v16h bhr = *(const v16h*)(WhL + ((c * 24 +      t) * 32 + lane) * 16);
      v16h bhz = *(const v16h*)(WhL + ((c * 24 +  8 + t) * 32 + lane) * 16);
      v16h bhn = *(const v16h*)(WhL + ((c * 24 + 16 + t) * 32 + lane) * 16);
      ir  = __builtin_amdgcn_wmma_f32_16x16x32_f16(false, aA, false, bir, (short)0, ir,  false, false);
      iz  = __builtin_amdgcn_wmma_f32_16x16x32_f16(false, aA, false, biz, (short)0, iz,  false, false);
      inn = __builtin_amdgcn_wmma_f32_16x16x32_f16(false, aA, false, bin, (short)0, inn, false, false);
      hr  = __builtin_amdgcn_wmma_f32_16x16x32_f16(false, aH, false, bhr, (short)0, hr,  false, false);
      hz  = __builtin_amdgcn_wmma_f32_16x16x32_f16(false, aH, false, bhz, (short)0, hz,  false, false);
      hnn = __builtin_amdgcn_wmma_f32_16x16x32_f16(false, aH, false, bhn, (short)0, hnn, false, false);
    }
    const int col = t * 16 + nbase;
    const float bir_ = gru_bi[col], biz_ = gru_bi[128 + col], bin_ = gru_bi[256 + col];
    const float bhr_ = gru_bh[col], bhz_ = gru_bh[128 + col], bhn_ = gru_bh[256 + col];
#pragma unroll
    for (int r = 0; r < 8; ++r) {
      const size_t row = (size_t)(nb + mbase + r);
      const float h  = node_feat[row * D_DIM + col];
      const float rg = 1.f / (1.f + __expf(-((ir[r] + bir_) + (hr[r] + bhr_))));
      const float zg = 1.f / (1.f + __expf(-((iz[r] + biz_) + (hz[r] + bhz_))));
      const float nx = (inn[r] + bin_) + rg * (hnn[r] + bhn_);
      const float e2 = __expf(2.f * nx);
      const float nn = (e2 - 1.f) / (e2 + 1.f);   // tanh
      out[row * D_DIM + col] = (1.f - zg) * nn + zg * h;
    }
  }
}

// ---------------------------------------------------------------------------
extern "C" void kernel_launch(void* const* d_in, const int* in_sizes, int n_in,
                              void* d_out, int out_size, void* d_ws, size_t ws_size,
                              hipStream_t stream) {
  const float* node_feat  = (const float*)d_in[0];
  const int*   edge_index = (const int*)  d_in[1];
  const float* edge_feat  = (const float*)d_in[2];
  const float* msg_W1     = (const float*)d_in[3];
  const float* msg_b1     = (const float*)d_in[4];
  const float* msg_W2     = (const float*)d_in[5];
  const float* msg_b2     = (const float*)d_in[6];
  const float* att_W1     = (const float*)d_in[7];
  const float* att_b1     = (const float*)d_in[8];
  const float* att_W2     = (const float*)d_in[9];
  const float* att_b2     = (const float*)d_in[10];
  const float* gru_Wi     = (const float*)d_in[11];
  const float* gru_Wh     = (const float*)d_in[12];
  const float* gru_bi     = (const float*)d_in[13];
  const float* gru_bh     = (const float*)d_in[14];
  (void)n_in; (void)out_size; (void)ws_size;

  const int Nn = in_sizes[0] / D_DIM;   // 50000 nodes
  const int Me = in_sizes[2] / E_DIM;   // 800000 edges

  // workspace layout; edge packs [mW1|aW1|mW2|aW2] and GRU packs [Wi|Wh] are
  // each contiguous so kernels copy them to LDS with one loop.
  char* ws = (char*)d_ws;
  size_t off = 0;
  float* accum = (float*)(ws + off); off += (size_t)Nn * D_DIM * sizeof(float);
  _Float16* mW1p = (_Float16*)(ws + off); off += (size_t)W1_HALFS * 2;
  _Float16* aW1p = (_Float16*)(ws + off); off += (size_t)W1_HALFS * 2;
  _Float16* mW2p = (_Float16*)(ws + off); off += (size_t)W2_HALFS * 2;
  _Float16* aW2p = (_Float16*)(ws + off); off += (size_t)W2_HALFS * 2;
  _Float16* Wip  = (_Float16*)(ws + off); off += (size_t)WG_HALFS * 2;
  _Float16* Whp  = (_Float16*)(ws + off); off += (size_t)WG_HALFS * 2;

  // 1) pack weights into WMMA B-fragment layout (f32 -> f16)
  gatgru_pack_w<<<(W1_HALFS + 255) / 256, 256, 0, stream>>>(msg_W1, mW1p, 5, 8, 128);
  gatgru_pack_w<<<(W1_HALFS + 255) / 256, 256, 0, stream>>>(att_W1, aW1p, 5, 8, 128);
  gatgru_pack_w<<<(W2_HALFS + 255) / 256, 256, 0, stream>>>(msg_W2, mW2p, 4, 8, 128);
  gatgru_pack_w<<<(W2_HALFS + 255) / 256, 256, 0, stream>>>(att_W2, aW2p, 4, 8, 128);
  gatgru_pack_w<<<(WG_HALFS + 255) / 256, 256, 0, stream>>>(gru_Wi, Wip, 4, 24, 384);
  gatgru_pack_w<<<(WG_HALFS + 255) / 256, 256, 0, stream>>>(gru_Wh, Whp, 4, 24, 384);

  // 2) zero the scatter accumulator (required each launch)
  const size_t acc4 = (size_t)Nn * D_DIM / 4;
  gatgru_zero<<<(int)((acc4 + 255) / 256), 256, 0, stream>>>(accum, acc4);

  // 3) fused edge pass: gather -> MLPs (WMMA) -> gate -> atomic scatter-add
  const int eblocks = (Me + 127) / 128;   // 8 waves/block x 16 edges/wave
  gatgru_edge_kernel<<<eblocks, 256, 0, stream>>>(
      node_feat, edge_index, edge_feat, msg_b1, msg_b2, att_b1, att_b2,
      mW1p, accum, Me);

  // 4) GRU update (WMMA) -> d_out
  const int gblocks = (Nn + 127) / 128;   // 8 waves/block x 16 nodes/wave
  gatgru_gru_kernel<<<gblocks, 256, 0, stream>>>(
      accum, node_feat, gru_bi, gru_bh, Wip, (float*)d_out, Nn);
}
